// RoPEMultiheadAttention_12859132084558
// MI455X (gfx1250) — matrix operand order repaired and checked by reading, not compiled, not run on hardware
//
#include <hip/hip_runtime.h>
#include <math.h>

// ---------------------------------------------------------------------------
// RoPE multi-head causal attention, MI455X (gfx1250, wave32).
// Matmuls: v_wmma_f32_16x16x32_bf16 (fp32 accumulate), bf16 operands
// pre-converted once. Projection GEMMs stage tiles into LDS with the CDNA5
// async path (global_load_async_to_lds_b128 + s_wait_asynccnt), double
// buffered so the copy of K-step n+1 overlaps the WMMAs of step n.
//
// Workspace (64 MiB peak, aliased; each region's fp32 producer is fully
// consumed before its bf16 consumer overwrites it):
//   [ 0,16M): q_f32            -> later kb  (bf16, after rope_q consumed q_f32)
//   [16,32M): k_f32            -> later vtb (bf16, after rope_k consumed k_f32)
//   [32,48M): v_f32            -> later aob (bf16, after v-transpose consumed v_f32)
//   [48,56M): xb (bf16)        -> later qb  (bf16, after the 3 projection GEMMs)
//   [56,64M): wqb,wkb,wvb,wob (bf16, 2 MiB each)
// ---------------------------------------------------------------------------

#define D_MODEL 1024
#define SEQ     1024
#define BATCH   4
#define NHEAD   16
#define DK      64
#define MROWS   (BATCH * SEQ)          // 4096
#define SCALE   0.125f                 // 1/sqrt(64)
#define LROW    40                     // padded LDS row stride (bf16 elems)

typedef __bf16 bf16_t;
typedef __attribute__((ext_vector_type(16))) __bf16 v16bf;
typedef __attribute__((ext_vector_type(8)))  __bf16 v8bf;
typedef __attribute__((ext_vector_type(4)))  __bf16 v4bf;
typedef __attribute__((ext_vector_type(8)))  float  v8f;

static __device__ __forceinline__ v8f wmma_bf16(v16bf a, v16bf b, v8f c) {
  return __builtin_amdgcn_wmma_f32_16x16x32_bf16(
      /*neg_a=*/false, a, /*neg_b=*/false, b,
      /*c_mod=*/(short)0, c, /*reuse_a=*/false, /*reuse_b=*/false);
}

static __device__ __forceinline__ v8f zero8() {
  v8f z = {0.f, 0.f, 0.f, 0.f, 0.f, 0.f, 0.f, 0.f};
  return z;
}

static __device__ __forceinline__ v16bf combine16(v8bf lo, v8bf hi) {
  union { v16bf v; v8bf h[2]; } u;
  u.h[0] = lo;
  u.h[1] = hi;
  return u.v;
}

// A fragment (16x32 bf16) from row-major bf16; per ISA 7.12.2 each lane's K set
// is {kb..kb+7} U {16+kb..16+kb+7} (kb = 0 / 8 by lane half) -> two 16B loads.
static __device__ __forceinline__ v16bf a_frag(const bf16_t* __restrict__ A,
                                               int lda, int row, int k0, int lane) {
  const int kb = (lane < 16) ? 0 : 8;
  const bf16_t* p = A + (size_t)row * lda + k0 + kb;
  return combine16(*(const v8bf*)p, *(const v8bf*)(p + 16));
}

// B fragment (32x16 bf16) with B[k][n] = M[nrow][k]; lane holds one n (nrow is
// passed per-lane), K set {kb2..kb2+7} U {16+kb2..16+kb2+7} (kb2 = 0 / 8) -> two 16B loads.
static __device__ __forceinline__ v16bf b_frag(const bf16_t* __restrict__ M,
                                               int ldm, int nrow, int k0, int lane) {
  const int kb2 = (lane < 16) ? 0 : 8;
  const bf16_t* p = M + (size_t)nrow * ldm + k0 + kb2;
  return combine16(*(const v8bf*)p, *(const v8bf*)(p + 16));
}

// CDNA5 async copy: 16B global -> LDS, tracked by ASYNCcnt (no VGPR round-trip).
// Generic pointers to LDS hold the LDS byte offset in their low 32 bits
// (ISA 10.2).
static __device__ __forceinline__ void async_b128(const bf16_t* g, const bf16_t* l) {
  const unsigned lofs = (unsigned)(size_t)l;
  asm volatile("global_load_async_to_lds_b128 %0, %1, off"
               :: "v"(lofs), "v"((unsigned long long)(size_t)g)
               : "memory");
}

static __device__ __forceinline__ float half_reduce_max(float v) {
#pragma unroll
  for (int off = 1; off < 16; off <<= 1)
    v = fmaxf(v, __shfl_xor(v, off, 32));
  return v;
}

static __device__ __forceinline__ float half_reduce_add(float v) {
#pragma unroll
  for (int off = 1; off < 16; off <<= 1)
    v += __shfl_xor(v, off, 32);
  return v;
}

// ---------------------------------------------------------------------------
// fp32 -> bf16 bulk convert (one float4 -> one 8B bf16x4 store per thread)
// ---------------------------------------------------------------------------
__global__ __launch_bounds__(256) void cvt_f32_bf16(const float* __restrict__ s,
                                                    bf16_t* __restrict__ d) {
  const int i = blockIdx.x * blockDim.x + threadIdx.x;
  const float4 f = ((const float4*)s)[i];
  v4bf o;
  o[0] = (bf16_t)f.x; o[1] = (bf16_t)f.y; o[2] = (bf16_t)f.z; o[3] = (bf16_t)f.w;
  ((v4bf*)d)[i] = o;
}

// ---------------------------------------------------------------------------
// Y[m][n] = sum_k X[m][k]*W[n][k], X/W bf16 row-major, Y fp32.
// 256 thr = 8 waves (2x4); block tile 64(M) x 128(N); K-step 32.
// Tiles staged to LDS with async b128 copies, double buffered:
//   issue(n+1) -> s_wait_asynccnt 3 (buffer n arrived; async loads complete
//   in order) -> barrier -> ds_load fragments + 4x WMMA -> barrier.
// X tile read by 4 waves, W tile by 2 -> 2-4x less L2 fetch traffic.
// ---------------------------------------------------------------------------
__global__ __launch_bounds__(256) void gemm_bf16(const bf16_t* __restrict__ X,
                                                 const bf16_t* __restrict__ W,
                                                 float* __restrict__ Y) {
  __shared__ __align__(16) bf16_t Xs[2][64 * LROW];    // 10 KiB
  __shared__ __align__(16) bf16_t Ws[2][128 * LROW];   // 20 KiB

  const int tid  = threadIdx.x;
  const int lane = tid & 31;
  const int wave = tid >> 5;
  const int l16  = lane & 15;
  const int mblk = blockIdx.x * 64;
  const int nblk = blockIdx.y * 128;
  const int mw = (wave >> 2) * 32;   // wave tile offset in block tile
  const int nw = (wave & 3) * 32;

  // Per-thread async-copy assignments (16B chunks of 8 bf16):
  // X tile 64x32 = 256 chunks (1/thread); W tile 128x32 = 512 chunks (2/thread).
  const int xr = tid >> 2, xc = (tid & 3) * 8;
  const int wr0 = tid >> 2,          wc0 = (tid & 3) * 8;
  const int wr1 = (tid + 256) >> 2,  wc1 = wc0;

  auto issue = [&](int buf, int kk) {
    async_b128(X + (size_t)(mblk + xr) * D_MODEL + kk + xc,
               &Xs[buf][xr * LROW + xc]);
    async_b128(W + (size_t)(nblk + wr0) * D_MODEL + kk + wc0,
               &Ws[buf][wr0 * LROW + wc0]);
    async_b128(W + (size_t)(nblk + wr1) * D_MODEL + kk + wc1,
               &Ws[buf][wr1 * LROW + wc1]);
  };

  v8f acc00 = zero8(), acc01 = zero8(), acc10 = zero8(), acc11 = zero8();

  issue(0, 0);
  const int NSTEP = D_MODEL / 32;
  for (int n = 0; n < NSTEP; ++n) {
    if (n + 1 < NSTEP) {
      issue((n + 1) & 1, (n + 1) * 32);
      asm volatile("s_wait_asynccnt 0x3" ::: "memory");  // buffer n complete
    } else {
      asm volatile("s_wait_asynccnt 0x0" ::: "memory");
    }
    __syncthreads();                                     // tiles visible to all waves

    const bf16_t* xs = &Xs[n & 1][0];
    const bf16_t* ws = &Ws[n & 1][0];
    const v16bf a0 = a_frag(xs, LROW, mw + l16,      0, lane);
    const v16bf a1 = a_frag(xs, LROW, mw + 16 + l16, 0, lane);
    const v16bf b0 = b_frag(ws, LROW, nw + l16,      0, lane);
    const v16bf b1 = b_frag(ws, LROW, nw + 16 + l16, 0, lane);
    acc00 = wmma_bf16(a0, b0, acc00);
    acc01 = wmma_bf16(a0, b1, acc01);
    acc10 = wmma_bf16(a1, b0, acc10);
    acc11 = wmma_bf16(a1, b1, acc11);

    __syncthreads();                                     // reads done before overwrite
  }

  const int m0 = mblk + mw, n0 = nblk + nw;
  const int half8 = (lane < 16) ? 0 : 8;
#pragma unroll
  for (int r = 0; r < 8; ++r) {
    const int r0 = m0 + half8 + r;
    const int r1 = m0 + 16 + half8 + r;
    Y[(size_t)r0 * D_MODEL + n0 + l16]      = acc00[r];
    Y[(size_t)r0 * D_MODEL + n0 + 16 + l16] = acc01[r];
    Y[(size_t)r1 * D_MODEL + n0 + l16]      = acc10[r];
    Y[(size_t)r1 * D_MODEL + n0 + 16 + l16] = acc11[r];
  }
}

// ---------------------------------------------------------------------------
// RoPE (fp32 math) + convert to bf16. One thread per (even,odd) pair.
// Reference ignores token_positions: pos = s = row % SEQ.
// ---------------------------------------------------------------------------
__global__ __launch_bounds__(256) void rope_f32_to_bf16(const float* __restrict__ src,
                                                        bf16_t* __restrict__ dst) {
  const int idx = blockIdx.x * blockDim.x + threadIdx.x;   // [0, 2M)
  const int row = idx >> 9;                                // 512 pairs per row
  const int pr  = idx & 511;
  const int head = pr >> 5;
  const int i    = pr & 31;
  const int col  = head * DK + 2 * i;

  const float* sp = src + (size_t)row * D_MODEL + col;
  const int pos = row & (SEQ - 1);
  const float freq = __expf(-0.28782313662425572f * (float)i);  // ln(1e4)/32
  float sn, cs;
  sincosf((float)pos * freq, &sn, &cs);
  const float xe = sp[0], xo = sp[1];

  union { bf16_t b[2]; unsigned int u; } o;
  o.b[0] = (bf16_t)(xe * cs - xo * sn);
  o.b[1] = (bf16_t)(xe * sn + xo * cs);
  *(unsigned int*)(dst + (size_t)row * D_MODEL + col) = o.u;
}

// ---------------------------------------------------------------------------
// vt[b][e][s] = (bf16) v[(b*SEQ+s)*D_MODEL + e]  (per-batch transpose)
// Makes the P*V B-fragment two contiguous 16B loads instead of a strided gather.
// ---------------------------------------------------------------------------
__global__ __launch_bounds__(256) void v_transpose_bf16(const float* __restrict__ v,
                                                        bf16_t* __restrict__ vt) {
  const int idx = blockIdx.x * blockDim.x + threadIdx.x;   // [0, 4M)
  const int s = idx & (SEQ - 1);
  const int e = (idx >> 10) & (D_MODEL - 1);
  const int b = idx >> 20;
  vt[idx] = (bf16_t)v[((size_t)b * SEQ + s) * D_MODEL + e];
}

// ---------------------------------------------------------------------------
// Flash attention: one wave per (b, h, 16-query tile); 32 keys per step.
// 4x WMMA QK^T, online softmax (16-lane shuffle reductions), P via padded
// wave-private LDS (C-frag -> A-frag reshape), 4x WMMA P*V. Output in bf16.
// ---------------------------------------------------------------------------
__global__ __launch_bounds__(128) void attn(const bf16_t* __restrict__ Q,
                                            const bf16_t* __restrict__ Km,
                                            const bf16_t* __restrict__ Vt,
                                            bf16_t* __restrict__ O) {
  __shared__ __align__(16) float ptile[4][16 * 34];  // stride 34: bank-spread

  const int lane = threadIdx.x & 31;
  const int wave = threadIdx.x >> 5;
  const int l16  = lane & 15;
  const int half8 = (lane < 16) ? 0 : 8;

  const int wid = blockIdx.x * 4 + wave;   // [0, 4096)
  const int qt  = wid & 63;
  const int bh  = wid >> 6;
  const int b   = bh >> 4;
  const int h   = bh & 15;
  const int q0  = qt * 16;

  const bf16_t* Qb  = Q  + (size_t)b * SEQ * D_MODEL;
  const bf16_t* Kb  = Km + (size_t)b * SEQ * D_MODEL;
  const bf16_t* Vtb = Vt + (size_t)b * D_MODEL * SEQ;
  bf16_t*       Ob  = O  + (size_t)b * SEQ * D_MODEL;

  const v16bf aq0 = a_frag(Qb, D_MODEL, q0 + l16, h * DK + 0,  lane);
  const v16bf aq1 = a_frag(Qb, D_MODEL, q0 + l16, h * DK + 32, lane);

  v8f acc0 = zero8(), acc1 = zero8(), acc2 = zero8(), acc3 = zero8();
  float mrow[8], lrow[8];
#pragma unroll
  for (int r = 0; r < 8; ++r) { mrow[r] = -INFINITY; lrow[r] = 0.f; }

  const int kend = q0 + 16;                 // causal: keys < kend
  for (int kt = 0; kt < kend; kt += 32) {
    v8f c0 = zero8(), c1 = zero8();
    {
      const v16bf bk00 = b_frag(Kb, D_MODEL, kt + l16,      h * DK + 0,  lane);
      const v16bf bk10 = b_frag(Kb, D_MODEL, kt + 16 + l16, h * DK + 0,  lane);
      const v16bf bk01 = b_frag(Kb, D_MODEL, kt + l16,      h * DK + 32, lane);
      const v16bf bk11 = b_frag(Kb, D_MODEL, kt + 16 + l16, h * DK + 32, lane);
      c0 = wmma_bf16(aq0, bk00, c0);
      c0 = wmma_bf16(aq1, bk01, c0);
      c1 = wmma_bf16(aq0, bk10, c1);
      c1 = wmma_bf16(aq1, bk11, c1);
    }

    const int col0 = kt + l16;
#pragma unroll
    for (int r = 0; r < 8; ++r) {
      const int row = q0 + half8 + r;
      const float s0 = (col0 <= row)      ? c0[r] * SCALE : -INFINITY;
      const float s1 = (col0 + 16 <= row) ? c1[r] * SCALE : -INFINITY;
      const float mx   = half_reduce_max(fmaxf(s0, s1));
      const float mnew = fmaxf(mrow[r], mx);
      const float alpha = __expf(mrow[r] - mnew);   // 0 on first slab
      const float e0 = __expf(s0 - mnew);           // masked -> 0
      const float e1 = __expf(s1 - mnew);
      const float rsum = half_reduce_add(e0 + e1);
      lrow[r] = lrow[r] * alpha + rsum;
      mrow[r] = mnew;
      acc0[r] *= alpha; acc1[r] *= alpha; acc2[r] *= alpha; acc3[r] *= alpha;
      ptile[wave][(half8 + r) * 34 + l16]      = e0;
      ptile[wave][(half8 + r) * 34 + l16 + 16] = e1;
    }

    // P (16x32) A-fragment from LDS (same-wave DS ops are in-order)
    v16bf ap;
    {
      const int kb = (lane < 16) ? 0 : 8;
      const int rowoff = l16 * 34;
#pragma unroll
      for (int j = 0; j < 8; ++j) {
        const int koff = (j < 4) ? (kb + 2 * j) : (16 + kb + 2 * (j - 4));
        ap[2 * j]     = (bf16_t)ptile[wave][rowoff + koff];
        ap[2 * j + 1] = (bf16_t)ptile[wave][rowoff + koff + 1];
      }
    }

    // P * V via transposed V: B[k][n] = Vt[h*64+f*16+n][kt+k], contiguous
    {
      const v16bf bv0 = b_frag(Vtb, SEQ, h * DK + 0  + l16, kt, lane);
      const v16bf bv1 = b_frag(Vtb, SEQ, h * DK + 16 + l16, kt, lane);
      const v16bf bv2 = b_frag(Vtb, SEQ, h * DK + 32 + l16, kt, lane);
      const v16bf bv3 = b_frag(Vtb, SEQ, h * DK + 48 + l16, kt, lane);
      acc0 = wmma_bf16(ap, bv0, acc0);
      acc1 = wmma_bf16(ap, bv1, acc1);
      acc2 = wmma_bf16(ap, bv2, acc2);
      acc3 = wmma_bf16(ap, bv3, acc3);
    }
  }

#pragma unroll
  for (int r = 0; r < 8; ++r) {
    const int row = q0 + half8 + r;
    const float inv = 1.0f / lrow[r];
    bf16_t* op = Ob + (size_t)row * D_MODEL + h * DK + l16;
    op[0]  = (bf16_t)(acc0[r] * inv);
    op[16] = (bf16_t)(acc1[r] * inv);
    op[32] = (bf16_t)(acc2[r] * inv);
    op[48] = (bf16_t)(acc3[r] * inv);
  }
}

// ---------------------------------------------------------------------------
extern "C" void kernel_launch(void* const* d_in, const int* in_sizes, int n_in,
                              void* d_out, int out_size, void* d_ws, size_t ws_size,
                              hipStream_t stream) {
  const float* x  = (const float*)d_in[0];
  // d_in[1] = token_positions (unused: reference RoPE uses arange(s))
  const float* Wq = (const float*)d_in[2];
  const float* Wk = (const float*)d_in[3];
  const float* Wv = (const float*)d_in[4];
  const float* Wo = (const float*)d_in[5];
  float* out = (float*)d_out;

  char* ws = (char*)d_ws;
  float*  qf  = (float*)(ws + (size_t)0);          // 16 MiB
  float*  kf  = (float*)(ws + ((size_t)16 << 20)); // 16 MiB
  float*  vf  = (float*)(ws + ((size_t)32 << 20)); // 16 MiB
  bf16_t* xb  = (bf16_t*)(ws + ((size_t)48 << 20));// 8 MiB
  bf16_t* wqb = (bf16_t*)(ws + ((size_t)56 << 20));// 2 MiB
  bf16_t* wkb = (bf16_t*)(ws + ((size_t)58 << 20));
  bf16_t* wvb = (bf16_t*)(ws + ((size_t)60 << 20));
  bf16_t* wob = (bf16_t*)(ws + ((size_t)62 << 20));
  // Aliased (producers fully consumed before overwrite -- see header comment):
  bf16_t* qb  = (bf16_t*)(ws + ((size_t)48 << 20));// over xb
  bf16_t* kb  = (bf16_t*)(ws + (size_t)0);         // over qf
  bf16_t* vtb = (bf16_t*)(ws + ((size_t)16 << 20));// over kf
  bf16_t* aob = (bf16_t*)(ws + ((size_t)32 << 20));// over vf

  // 1) one-time bf16 conversion of activations + weights
  cvt_f32_bf16<<<(MROWS * D_MODEL / 4) / 256, 256, 0, stream>>>(x, xb);
  cvt_f32_bf16<<<(D_MODEL * D_MODEL / 4) / 256, 256, 0, stream>>>(Wq, wqb);
  cvt_f32_bf16<<<(D_MODEL * D_MODEL / 4) / 256, 256, 0, stream>>>(Wk, wkb);
  cvt_f32_bf16<<<(D_MODEL * D_MODEL / 4) / 256, 256, 0, stream>>>(Wv, wvb);
  cvt_f32_bf16<<<(D_MODEL * D_MODEL / 4) / 256, 256, 0, stream>>>(Wo, wob);

  // 2) Q/K/V projections (bf16 WMMA, async-LDS staged, fp32 out)
  const dim3 gblk(MROWS / 64, D_MODEL / 128, 1);   // 64 x 8
  gemm_bf16<<<gblk, 256, 0, stream>>>(xb, wqb, qf);
  gemm_bf16<<<gblk, 256, 0, stream>>>(xb, wkb, kf);
  gemm_bf16<<<gblk, 256, 0, stream>>>(xb, wvb, vf);

  // 3) RoPE (fp32) -> bf16 operands; V -> per-batch transposed bf16
  rope_f32_to_bf16<<<(MROWS * 512) / 256, 256, 0, stream>>>(qf, qb);
  rope_f32_to_bf16<<<(MROWS * 512) / 256, 256, 0, stream>>>(kf, kb);
  v_transpose_bf16<<<(MROWS * D_MODEL) / 256, 256, 0, stream>>>(vf, vtb);

  // 4) causal flash attention (bf16 WMMA, fp32 softmax), bf16 out
  attn<<<(BATCH * NHEAD * (SEQ / 16)) / 4, 128, 0, stream>>>(qb, kb, vtb, aob);

  // 5) output projection -> fp32 result
  gemm_bf16<<<gblk, 256, 0, stream>>>(aob, wob, out);
}
